// DoubleAlignConv_26491358282333
// MI455X (gfx1250) — matrix operand order repaired and checked by reading, not compiled, not run on hardware
//
#include <hip/hip_runtime.h>
#include <hip/hip_bf16.h>

// ---------------- problem constants ----------------
#define BSZ   2
#define CCH   256
#define OCH   256
#define HIM   64
#define WIM   64
#define NPIX  (HIM * WIM)      // 4096
#define KPTS  9                // 3x3

// ---------------- GEMM tiling ----------------------
#define BM    256              // ALL output channels per block -> each sample gathered once
#define BN    32               // pixels per block
#define BK    32               // K-slice = 32 channels @ one kernel point
#define LDK   40               // padded LDS row stride in halves (80B, 16B-aligned)
#define NSLICE ((CCH / BK) * KPTS)   // 72 K-slices

typedef __attribute__((ext_vector_type(16))) _Float16 v16h;
typedef __attribute__((ext_vector_type(8)))  float    v8f;

struct Coord {                 // bilinear taps, validity folded into weights
    float w00, w01, w10, w11;
    int   o00, o01, o10, o11;  // clamped flat element offsets y*W+x (non-negative)
};

// ============================================================
// Kernel 1: absolute sampling coordinates per (variant,b,pixel,r)
//   v=0 : feature-refined offsets (w_irr -> x_feat, output first)
//   v=1 : anchor offsets          (w_grid -> x_anchor)
// The reference's "offset + grid re-add" collapses to R(a)*p + ctr.
// ============================================================
__global__ void coords_kernel(const float* __restrict__ anchors,
                              const float* __restrict__ feats,
                              float* __restrict__ cws) {
    int idx = blockIdx.x * blockDim.x + threadIdx.x;   // b*4096 + p
    if (idx >= BSZ * NPIX) return;
    const float* a = anchors + (size_t)idx * 5;
    float xc = a[0] * 0.125f, yc = a[1] * 0.125f;
    float w  = a[2] * 0.125f, h  = a[3] * 0.125f;
    float s, c;
    __sincosf(a[4], &s, &c);
    float dw = w * (1.0f / 3.0f), dh = h * (1.0f / 3.0f);
    const float* f = feats + (size_t)idx * (2 * KPTS);
    int b = idx >> 12, p = idx & (NPIX - 1);

    for (int r = 0; r < KPTS; ++r) {
        float kx = (float)(r % 3) - 1.0f;
        float ky = (float)(r / 3) - 1.0f;
        // v = 1 : anchor grid
        {
            float pxi = dw * kx, pyi = dh * ky;
            float px = c * pxi - s * pyi + xc;
            float py = s * pxi + c * pyi + yc;
            float* o = cws + ((((size_t)BSZ + b) * NPIX + p) * KPTS + r) * 2;
            o[0] = py; o[1] = px;
        }
        // v = 0 : feature-refined
        {
            float pxi = dw * (kx - 0.5f + f[r]);
            float pyi = dh * (ky - 0.5f + f[KPTS + r]);
            float px = c * pxi - s * pyi + xc;
            float py = s * pxi + c * pyi + yc;
            float* o = cws + (((size_t)b * NPIX + p) * KPTS + r) * 2;
            o[0] = py; o[1] = px;
        }
    }
}

// ============================================================
// Kernel 2: pack weights -> half, layout wpk[v][r][m][c].
// A 64B row == one WMMA A-fragment row, so fragments are loaded
// DIRECTLY from global (2x global_load_b128 per lane), no LDS.
// ============================================================
__global__ void packw_kernel(const float* __restrict__ w_grid,
                             const float* __restrict__ w_irr,
                             _Float16* __restrict__ wpk) {
    int idx = blockIdx.x * blockDim.x + threadIdx.x;   // ((v*9+r)*256 + m)*256 + c
    if (idx >= 2 * KPTS * OCH * CCH) return;
    int c = idx & 255;
    int m = (idx >> 8) & 255;
    int t = idx >> 16;              // v*9 + r
    int r = t % KPTS, v = t / KPTS;
    const float* src = (v == 0) ? w_irr : w_grid;
    wpk[idx] = (_Float16)src[((size_t)m * CCH + c) * KPTS + r];
}

// ============================================================
// Kernel 3: fused bilinear gather + implicit GEMM via WMMA.
// Block = 256 thr = 8 wave32; wave wv owns rows [wv*32, wv*32+32)
// x all 32 pixels = 2x2 WMMA 16x16 tiles.
//  - A fragments: loaded straight from packed weights in global (L2).
//  - B tile (bilinear samples): double-buffered LDS; tap loads for
//    slice s+1 issue before the WMMAs of slice s, their FMA-combine
//    and ds_store_b64 happen after (loads-early / waits-late split).
// ============================================================
__global__ __launch_bounds__(256, 1)
void deform_wmma_kernel(const float* __restrict__ x,
                        const float* __restrict__ cws,
                        const _Float16* __restrict__ wpk,
                        float* __restrict__ out) {
    __shared__ __align__(16) _Float16 ldsB[2][BN * LDK];   // sampled [n][kk]
    __shared__ Coord scoord[BN * KPTS];                    // bilinear taps per (n,r)

    const int tid  = threadIdx.x;
    const int lane = tid & 31;
    const int wv   = tid >> 5;         // 0..7 : M-wave
    const int z    = blockIdx.z;       // v*BSZ + b
    const int b    = z & 1;
    const int v    = z >> 1;
    const int p0   = blockIdx.x * BN;  // pixel tile base
    const int hi   = (lane >= 16) ? 1 : 0;
    const int l15  = lane & 15;

    // ---- bilinear tap table: once per block, reused by all 72 K-slices ----
    for (int e = tid; e < BN * KPTS; e += 256) {
        int n = e / KPTS, r = e - n * KPTS;
        const float* cc = cws + (((size_t)z * NPIX + (p0 + n)) * KPTS + r) * 2;
        float py = cc[0], px = cc[1];
        float y0f = floorf(py), x0f = floorf(px);
        int   y0 = (int)y0f,   x0 = (int)x0f;
        int   y1 = y0 + 1,     x1 = x0 + 1;
        float wy = py - y0f,   wx = px - x0f;
        float vy0 = (y0 >= 0 && y0 < HIM) ? 1.f : 0.f;
        float vy1 = (y1 >= 0 && y1 < HIM) ? 1.f : 0.f;
        float vx0 = (x0 >= 0 && x0 < WIM) ? 1.f : 0.f;
        float vx1 = (x1 >= 0 && x1 < WIM) ? 1.f : 0.f;
        int cy0 = min(max(y0, 0), HIM - 1), cy1 = min(max(y1, 0), HIM - 1);
        int cx0 = min(max(x0, 0), WIM - 1), cx1 = min(max(x1, 0), WIM - 1);
        Coord cd;
        cd.w00 = (1.f - wy) * (1.f - wx) * vy0 * vx0;
        cd.w01 = (1.f - wy) * wx         * vy0 * vx1;
        cd.w10 = wy         * (1.f - wx) * vy1 * vx0;
        cd.w11 = wy         * wx         * vy1 * vx1;
        cd.o00 = cy0 * WIM + cx0;  cd.o01 = cy0 * WIM + cx1;
        cd.o10 = cy1 * WIM + cx0;  cd.o11 = cy1 * WIM + cx1;
        scoord[e] = cd;
    }
    __syncthreads();

    // sample staging: thread t stages pixel n = t&31, channels kg*4..kg*4+3
    const int nB  = tid & 31;
    const int kgB = tid >> 5;          // 0..7
    const float*    xb    = x   + (size_t)b * CCH * NPIX;   // uniform base (SADDR)
    const _Float16* wbase = wpk + (size_t)v * KPTS * OCH * CCH;

    v8f acc[2][2] = {};

    // ---- prologue: fully stage slice 0 into ldsB[0] ----
    {
        Coord cd = scoord[nB * KPTS + 0];
        int base_i = (kgB << 2) * NPIX;
        int i00 = base_i + cd.o00, i01 = base_i + cd.o01;
        int i10 = base_i + cd.o10, i11 = base_i + cd.o11;
        union { _Float16 h[4]; uint2 u; } pk;
        #pragma unroll
        for (int i = 0; i < 4; ++i) {
            int di = i * NPIX;
            pk.h[i] = (_Float16)(cd.w00 * xb[i00 + di] + cd.w01 * xb[i01 + di]
                               + cd.w10 * xb[i10 + di] + cd.w11 * xb[i11 + di]);
        }
        *(uint2*)(&ldsB[0][nB * LDK + kgB * 4]) = pk.u;
    }
    __syncthreads();

    int c0c = 0, rc = 0;                    // (c0, r) of the current slice
    for (int s = 0; s < NSLICE; ++s) {
        const int cur = s & 1;
        const bool have_next = (s + 1 < NSLICE);
        int c0n = c0c, rn = rc + 1;
        if (rn == KPTS) { rn = 0; c0n += BK; }

        // ---- issue next slice's 16 raw tap loads (no FMA yet -> no wait) ----
        Coord cd;
        float tap[16];
        if (have_next) {
            cd = scoord[nB * KPTS + rn];
            int base_i = (c0n + (kgB << 2)) * NPIX;
            int i00 = base_i + cd.o00, i01 = base_i + cd.o01;
            int i10 = base_i + cd.o10, i11 = base_i + cd.o11;
            #pragma unroll
            for (int i = 0; i < 4; ++i) {
                int di = i * NPIX;                  // compile-time -> imm offset
                tap[4 * i + 0] = xb[i00 + di];
                tap[4 * i + 1] = xb[i01 + di];
                tap[4 * i + 2] = xb[i10 + di];
                tap[4 * i + 3] = xb[i11 + di];
            }
        }

        // ---- A fragments straight from global packed weights ----
        union { uint4 q[2]; v16h h; } fa[2], fb[2];
        const _Float16* wrow = wbase + ((size_t)rc * OCH) * CCH + c0c;
        #pragma unroll
        for (int tm = 0; tm < 2; ++tm) {
            const _Float16* pA = wrow + (size_t)(wv * 32 + tm * 16 + l15) * CCH;
            fa[tm].q[0] = *(const uint4*)(pA + hi * 8);       // K 0..7  (+8 hi-half)
            fa[tm].q[1] = *(const uint4*)(pA + 16 + hi * 8);  // K 16..23(+8 hi-half)
        }
        // ---- B fragments from LDS ----
        #pragma unroll
        for (int tn = 0; tn < 2; ++tn) {
            const _Float16* pB = &ldsB[cur][(tn * 16 + l15) * LDK];
            fb[tn].q[0] = *(const uint4*)(pB + hi * 8);       // K 0..7  (+8 hi-half)
            fb[tn].q[1] = *(const uint4*)(pB + 16 + hi * 8);  // K 16..23(+8 hi-half)
        }

        #pragma unroll
        for (int tm = 0; tm < 2; ++tm)
            #pragma unroll
            for (int tn = 0; tn < 2; ++tn)
                acc[tm][tn] = __builtin_amdgcn_wmma_f32_16x16x32_f16(
                    false, fa[tm].h, false, fb[tn].h,
                    (short)0, acc[tm][tn], false, false);

        // ---- combine taps + store next B tile ----
        if (have_next) {
            union { _Float16 h[4]; uint2 u; } pk;
            #pragma unroll
            for (int i = 0; i < 4; ++i)
                pk.h[i] = (_Float16)(cd.w00 * tap[4 * i + 0] + cd.w01 * tap[4 * i + 1]
                                   + cd.w10 * tap[4 * i + 2] + cd.w11 * tap[4 * i + 3]);
            *(uint2*)(&ldsB[cur ^ 1][nB * LDK + kgB * 4]) = pk.u;
        }
        c0c = c0n; rc = rn;
        __syncthreads();                       // one barrier per slice
    }

    // ---- ReLU + store (C/D layout: vgpr j -> M=j+8*hi, N=l15) ----
    float* ob = out + (size_t)z * OCH * NPIX;
    #pragma unroll
    for (int tm = 0; tm < 2; ++tm)
        #pragma unroll
        for (int tn = 0; tn < 2; ++tn) {
            int col = p0 + tn * 16 + l15;
            #pragma unroll
            for (int j = 0; j < 8; ++j) {
                int row = wv * 32 + tm * 16 + j + hi * 8;
                float vv = acc[tm][tn][j];
                ob[(size_t)row * NPIX + col] = vv > 0.f ? vv : 0.f;
            }
        }
}

// ============================================================
extern "C" void kernel_launch(void* const* d_in, const int* in_sizes, int n_in,
                              void* d_out, int out_size, void* d_ws, size_t ws_size,
                              hipStream_t stream) {
    const float* x       = (const float*)d_in[0];
    const float* anchors = (const float*)d_in[1];
    const float* feats   = (const float*)d_in[2];
    const float* w_grid  = (const float*)d_in[3];
    const float* w_irr   = (const float*)d_in[4];
    float*       out     = (float*)d_out;

    // workspace: coords (2*B*4096*9*2 f32 = 1.125 MB) then packed half weights (2.25 MB)
    float*    cws = (float*)d_ws;
    _Float16* wpk = (_Float16*)((char*)d_ws + (size_t)2 * BSZ * NPIX * KPTS * 2 * sizeof(float));

    {
        int tot = BSZ * NPIX;
        coords_kernel<<<(tot + 255) / 256, 256, 0, stream>>>(anchors, feats, cws);
    }
    {
        int tot = 2 * KPTS * OCH * CCH;
        packw_kernel<<<(tot + 255) / 256, 256, 0, stream>>>(w_grid, w_irr, wpk);
    }
    {
        dim3 grid(NPIX / BN, 1, BSZ * 2);   // 128 x 1 x 4 blocks
        deform_wmma_kernel<<<grid, 256, 0, stream>>>(x, cws, wpk, out);
    }
}
